// LSTM_FEAT_24524263260274
// MI455X (gfx1250) — matrix operand order repaired and checked by reading, not compiled, not run on hardware
//
#include <hip/hip_runtime.h>
#include <cstdint>

// ---------------- problem constants ----------------
#define TSTEPS 2048
#define IN_DIM 568
#define EMBED  128
#define HID    1024
#define OUTD   256

#define NWG    128      // persistent workgroups, 1 per WGP (assumes >=128 WGPs resident)
#define NTHR   256      // 8 wave32 per WG

#define K0     1720     // IN_DIM + EMBED + HID  (layer0 fused input)
#define K0CAT  696      // IN_DIM + EMBED
#define NC0    54       // ceil(1720/32)
#define K1     2048     // HID + HID             (layer1 fused input)
#define NC1    64
#define NCLIN  32       // 1024/32
#define NCMAP  8        // 256/32

// ---------------- workspace layout (bytes) ----------------
static constexpr long long WS_BAR   = 0;                          // 2 x u32 barrier
static constexpr long long WS_H0    = 256;                        // 1024 f32
static constexpr long long WS_H1    = 4352;                       // 1024 f32
static constexpr long long WS_EMB   = 8448;                       // 128 f32
static constexpr long long WS_W0    = 16384;                      // packed bf16 frags
static constexpr long long SZ_W0_WG = 2ll * NC0 * 1024;           // 110592 B / WG
static constexpr long long WS_W1    = WS_W0 + (long long)NWG * SZ_W0_WG;
static constexpr long long SZ_W1_WG = 2ll * NC1 * 1024;           // 131072 B / WG
static constexpr long long WS_WLIN  = WS_W1 + (long long)NWG * SZ_W1_WG;
static constexpr long long SZ_WLIN  = 16ll * NCLIN * 1024;        // 512 KB
static constexpr long long WS_WMAP  = WS_WLIN + SZ_WLIN;
static constexpr long long SZ_WMAP  = 8ll * NCMAP * 1024;         // 64 KB
// total ~31.6 MB

// packed element counts (512 bf16 per 1KB fragment block)
static constexpr long long NP0 = (long long)NWG * 2 * NC0 * 512;
static constexpr long long NP1 = (long long)NWG * 2 * NC1 * 512;
static constexpr long long NPL = 16ll * NCLIN * 512;
static constexpr long long NPM = 8ll  * NCMAP * 512;

// ---------------- LDS layout (bytes) ----------------
static constexpr unsigned OFF_WL0  = 0;         // 110592
static constexpr unsigned OFF_WL1  = 110592;    // 131072
static constexpr unsigned OFF_V    = 241664;    // 4096  (2048 bf16 staged vector)
static constexpr unsigned OFF_ZRED = 245760;    // 128   (32 f32 cross-wave reduce)
static constexpr unsigned OFF_B0   = 245888;    // 128   (layer0 fused bias, 32 f32)
static constexpr unsigned OFF_B1   = 246016;    // 128
static constexpr unsigned OFF_CST  = 246144;    // 64    (c0[8], c1[8])
static constexpr unsigned OFF_Y    = 246208;    // 1024  (WG0: y[256])
static constexpr unsigned OFF_P    = 247232;    // 512   (WG0: softmax bf16[256])
static constexpr unsigned OFF_WRED = 247744;    // 128
static constexpr unsigned LDS_TOTAL = 247872;   // < 320KB

typedef __bf16 v16bf __attribute__((ext_vector_type(16)));
typedef float  v8f   __attribute__((ext_vector_type(8)));

// ---------------- TDM availability ----------------
#if defined(__HIP_DEVICE_COMPILE__) && defined(__has_builtin)
#if __has_builtin(__builtin_amdgcn_tensor_load_to_lds)
#define HAVE_TDM 1
#endif
#endif

#if defined(HAVE_TDM)
typedef unsigned int tdm_v4u __attribute__((ext_vector_type(4)));
typedef int          tdm_v8i __attribute__((ext_vector_type(8)));
typedef int          tdm_v4i __attribute__((ext_vector_type(4)));

// 1D TDM copy: nbytes (multiple of 8, nbytes/8 < 65536) global -> LDS.
// D# built per CDNA5 ISA ch.8: group0 = {flags, lds_addr, global_addr, type=2},
// group1 = {data_size=8B, tensor_dim0=tile_dim0=n8, dim1=1, stride0=n8}.
__device__ __forceinline__ void tdm_load_1d(const void* gsrc, unsigned lds_off, unsigned nbytes) {
  unsigned long long ga = (unsigned long long)(uintptr_t)gsrc;
  unsigned n8 = nbytes >> 3;
  tdm_v4u g0;
  g0.x = 1u;                                          // count=1, user mode
  g0.y = lds_off;                                     // lds_addr  [63:32]
  g0.z = (unsigned)ga;                                // gaddr lo  [95:64]
  g0.w = ((unsigned)(ga >> 32) & 0x1FFFFFFu) | (2u << 30);  // gaddr hi + type=2
  tdm_v8i g1;
  g1[0] = (int)(3u << 16);                            // data_size = 8B
  g1[1] = (int)((n8 & 0xFFFFu) << 16);                // tensor_dim0[15:0]
  g1[2] = (int)(((n8 >> 16) & 0xFFFFu) | (1u << 16)); // tensor_dim0[31:16] | tensor_dim1=1
  g1[3] = (int)((n8 & 0xFFFFu) << 16);                // tile_dim0 = n8
  g1[4] = 0;                                          // tile_dim1/2 unused
  g1[5] = (int)n8;                                    // tensor_dim0_stride lo
  g1[6] = 0;
  g1[7] = 0;
  tdm_v4i z4 = {0, 0, 0, 0};
#if __clang_major__ >= 23
  tdm_v8i z8 = {0, 0, 0, 0, 0, 0, 0, 0};
  __builtin_amdgcn_tensor_load_to_lds(g0, g1, z4, z4, z8, 0);
#else
  __builtin_amdgcn_tensor_load_to_lds(g0, g1, z4, z4, 0);
#endif
}
#endif

// ---------------- helpers ----------------
__device__ __forceinline__ v16bf splat16(__bf16 s) {
  v16bf r;
#pragma unroll
  for (int i = 0; i < 16; ++i) r[i] = s;
  return r;
}
__device__ __forceinline__ float sigf(float z) { return 1.f / (1.f + __expf(-z)); }

// one WMMA GEMV step: A fragment from `wbase` (pre-packed 1KB blocks), B = staged
// vector chunk in the same K order as A: half h holds K {8h..8h+7, 16+8h..16+8h+7}
// (same for every column N).
__device__ __forceinline__ v8f gemv_wmma(const unsigned char* wbase, const __bf16* vbuf,
                                         int blk, int c, int lane, v8f acc) {
  const v16bf a = *(const v16bf*)(wbase + (((blk + c) << 10) + (lane << 5)));
  const int hb = (c << 5) + ((lane >> 4) << 3);
  v16bf b;
#pragma unroll
  for (int s = 0; s < 8; ++s) { b[s] = vbuf[hb + s]; b[8 + s] = vbuf[hb + 16 + s]; }
  return __builtin_amdgcn_wmma_f32_16x16x32_bf16(false, a, false,
                                                 b,
                                                 (short)0, acc, false, false);
}

// dual-accumulator GEMV over chunks {part, part+4, ...} < nc, stride 4 between
// the 4 waves sharing a tile; two independent WMMA chains to avoid D->C RAW
// stalls (ISA 7.12.1), combined at the end.
__device__ __forceinline__ v8f gemv_tile(const unsigned char* wbase, const __bf16* vbuf,
                                         int tile, int nc, int part, int lane) {
  const int blk = tile * nc;
  v8f acc0 = {}, acc1 = {};
  int c = part;
  for (; c + 4 < nc; c += 8) {
    acc0 = gemv_wmma(wbase, vbuf, blk, c, lane, acc0);
    acc1 = gemv_wmma(wbase, vbuf, blk, c + 4, lane, acc1);
  }
  if (c < nc) acc0 = gemv_wmma(wbase, vbuf, blk, c, lane, acc0);
#pragma unroll
  for (int i = 0; i < 8; ++i) acc0[i] += acc1[i];
  return acc0;
}

// sense-reversing grid barrier, agent scope
__device__ __forceinline__ void grid_barrier(unsigned* bar) {
  __threadfence();
  __syncthreads();
  if (threadIdx.x == 0) {
    unsigned* cnt = bar;
    unsigned* gen = bar + 1;
    unsigned gv = __hip_atomic_load(gen, __ATOMIC_RELAXED, __HIP_MEMORY_SCOPE_AGENT);
    unsigned a  = __hip_atomic_fetch_add(cnt, 1u, __ATOMIC_ACQ_REL, __HIP_MEMORY_SCOPE_AGENT);
    if (a == (unsigned)NWG - 1u) {
      __hip_atomic_store(cnt, 0u, __ATOMIC_RELAXED, __HIP_MEMORY_SCOPE_AGENT);
      __hip_atomic_store(gen, gv + 1u, __ATOMIC_RELEASE, __HIP_MEMORY_SCOPE_AGENT);
    } else {
      while (__hip_atomic_load(gen, __ATOMIC_ACQUIRE, __HIP_MEMORY_SCOPE_AGENT) == gv)
        __builtin_amdgcn_s_sleep(2);
    }
  }
  __syncthreads();
}

// ================= prep kernel =================
// Zeroes state region and packs all weights into bf16 WMMA A-fragment order:
// block(1KB) = 32 lanes x 16 bf16; lane<16 holds M=lane, K in {0..7,16..23};
// lane>=16 holds M=lane-16, K in {8..15,24..31} (16-bit A layout, ISA 7.12.2).
// Row gather interleaves i/f gates in tile0 and g/o gates in tile1 so each WG
// owns complete gate quadruples for hidden indices [g*8, g*8+8).
__global__ void lstm_prep(const float* __restrict__ Wih0, const float* __restrict__ Whh0,
                          const float* __restrict__ Wih1, const float* __restrict__ Whh1,
                          const float* __restrict__ Wlin, const float* __restrict__ Wmap,
                          unsigned char* __restrict__ ws) {
  long long id = (long long)blockIdx.x * blockDim.x + threadIdx.x;
  if (id < 4096) ((unsigned*)ws)[id] = 0u;   // barrier + h0 + h1 + embed
  long long e = id;
  if (e < NP0) {
    int li = (int)(e & 511), blk = (int)(e >> 9);
    int lane = li >> 4, j = li & 15, m = lane & 15;
    int chunk = blk % NC0, tb = blk / NC0;
    int tile = tb & 1, wg = tb >> 1;
    int row = (tile * 2 + (m >> 3)) * HID + wg * 8 + (m & 7);
    int kin = (lane < 16) ? ((j < 8) ? j : j + 8) : ((j < 8) ? j + 8 : j + 16);
    int k = chunk * 32 + kin;
    float v = 0.f;
    if (k < K0CAT)    v = Wih0[(long long)row * K0CAT + k];
    else if (k < K0)  v = Whh0[(long long)row * HID + (k - K0CAT)];
    ((__bf16*)(ws + WS_W0))[e] = (__bf16)v;
    return;
  }
  e -= NP0;
  if (e < NP1) {
    int li = (int)(e & 511), blk = (int)(e >> 9);
    int lane = li >> 4, j = li & 15, m = lane & 15;
    int chunk = blk % NC1, tb = blk / NC1;
    int tile = tb & 1, wg = tb >> 1;
    int row = (tile * 2 + (m >> 3)) * HID + wg * 8 + (m & 7);
    int kin = (lane < 16) ? ((j < 8) ? j : j + 8) : ((j < 8) ? j + 8 : j + 16);
    int k = chunk * 32 + kin;
    float v = (k < HID) ? Wih1[(long long)row * HID + k]
                        : Whh1[(long long)row * HID + (k - HID)];
    ((__bf16*)(ws + WS_W1))[e] = (__bf16)v;
    return;
  }
  e -= NP1;
  if (e < NPL) {
    int li = (int)(e & 511), blk = (int)(e >> 9);
    int lane = li >> 4, j = li & 15, m = lane & 15;
    int chunk = blk % NCLIN, tau = blk / NCLIN;
    int row = tau * 16 + m;
    int kin = (lane < 16) ? ((j < 8) ? j : j + 8) : ((j < 8) ? j + 8 : j + 16);
    int k = chunk * 32 + kin;
    ((__bf16*)(ws + WS_WLIN))[e] = (__bf16)Wlin[(long long)row * HID + k];
    return;
  }
  e -= NPL;
  if (e < NPM) {
    int li = (int)(e & 511), blk = (int)(e >> 9);
    int lane = li >> 4, j = li & 15, m = lane & 15;
    int chunk = blk % NCMAP, tau = blk / NCMAP;
    int row = tau * 16 + m;
    int kin = (lane < 16) ? ((j < 8) ? j : j + 8) : ((j < 8) ? j + 8 : j + 16);
    int k = chunk * 32 + kin;
    ((__bf16*)(ws + WS_WMAP))[e] = (__bf16)Wmap[(long long)row * OUTD + k];
  }
}

// ================= persistent LSTM kernel =================
__global__ __launch_bounds__(NTHR, 1)
void lstm_persistent(const float* __restrict__ x,
                     const float* __restrict__ bih0, const float* __restrict__ bhh0,
                     const float* __restrict__ bih1, const float* __restrict__ bhh1,
                     const float* __restrict__ blin, const float* __restrict__ bmap,
                     unsigned char* __restrict__ ws, float* __restrict__ out) {
  extern __shared__ unsigned char smem[];
  const int tid = threadIdx.x, g = blockIdx.x;
  const int lane = tid & 31;
  // wave id / tile / K-partition are wave-uniform: pin them to SGPRs so WMMA
  // loops get scalar control flow (EXEC stays all-1s, ISA 7.12 requirement).
  const int w    = __builtin_amdgcn_readfirstlane(tid >> 5);
  const int tile = w >> 2, part = w & 3;

  __bf16* wl0  = (__bf16*)(smem + OFF_WL0);
  __bf16* wl1  = (__bf16*)(smem + OFF_WL1);
  __bf16* vbuf = (__bf16*)(smem + OFF_V);
  float*  zred = (float*) (smem + OFF_ZRED);
  float*  b0l  = (float*) (smem + OFF_B0);
  float*  b1l  = (float*) (smem + OFF_B1);
  float*  cst  = (float*) (smem + OFF_CST);
  float*  ybuf = (float*) (smem + OFF_Y);
  __bf16* pbuf = (__bf16*)(smem + OFF_P);
  float*  wred = (float*) (smem + OFF_WRED);

  unsigned* bar = (unsigned*)(ws + WS_BAR);
  float* h0g  = (float*)(ws + WS_H0);
  float* h1g  = (float*)(ws + WS_H1);
  float* embg = (float*)(ws + WS_EMB);

  // ---- load this WG's weight slices into LDS (once, via Tensor Data Mover) ----
#if defined(HAVE_TDM)
  if (w == 0) {
    tdm_load_1d(ws + WS_W0 + (long long)g * SZ_W0_WG, (unsigned)(uintptr_t)wl0, (unsigned)SZ_W0_WG);
    tdm_load_1d(ws + WS_W1 + (long long)g * SZ_W1_WG, (unsigned)(uintptr_t)wl1, (unsigned)SZ_W1_WG);
    __builtin_amdgcn_s_wait_tensorcnt(0);
  }
#else
  {
    const uint4* s0 = (const uint4*)(ws + WS_W0 + (long long)g * SZ_W0_WG);
    uint4* d0 = (uint4*)wl0;
    for (int i = tid; i < (int)(SZ_W0_WG / 16); i += NTHR) d0[i] = s0[i];
    const uint4* s1 = (const uint4*)(ws + WS_W1 + (long long)g * SZ_W1_WG);
    uint4* d1 = (uint4*)wl1;
    for (int i = tid; i < (int)(SZ_W1_WG / 16); i += NTHR) d1[i] = s1[i];
  }
#endif
  // fused biases for this WG's 8 hidden indices, c-state init
  if (tid < 8) {
    int jj = g * 8 + tid;
#pragma unroll
    for (int q = 0; q < 4; ++q) {
      b0l[q * 8 + tid] = bih0[q * HID + jj] + bhh0[q * HID + jj];
      b1l[q * 8 + tid] = bih1[q * HID + jj] + bhh1[q * HID + jj];
    }
  }
  if (tid < 16) cst[tid] = 0.f;
  __syncthreads();

  for (int t = 0; t < TSTEPS; ++t) {
    // ================= layer 0 =================
    const float* xt = x + (long long)t * IN_DIM;
    if (t + 1 < TSTEPS) __builtin_prefetch(xt + IN_DIM, 0, 1);  // next step's input row
    for (int i = tid; i < NC0 * 32; i += NTHR) {
      float v = 0.f;
      if (i < IN_DIM)           v = xt[i];
      else if (i < K0CAT)       v = embg[i - IN_DIM];
      else if (i < K0)          v = h0g[i - K0CAT];
      vbuf[i] = (__bf16)v;
    }
    if (tid < 32) zred[tid] = 0.f;
    __syncthreads();
    {
      v8f acc = gemv_tile((const unsigned char*)wl0, vbuf, tile, NC0, part, lane);
      if ((lane & 15) == 0) {
        int ro = tile * 16 + ((lane >> 4) << 3);
#pragma unroll
        for (int i = 0; i < 8; ++i) atomicAdd(&zred[ro + i], acc[i]);
      }
    }
    __syncthreads();
    if (tid < 8) {  // zred = [i(8) f(8) g(8) o(8)]
      float zi = zred[tid] + b0l[tid], zf = zred[8 + tid] + b0l[8 + tid];
      float zg = zred[16 + tid] + b0l[16 + tid], zo = zred[24 + tid] + b0l[24 + tid];
      float cn = sigf(zf) * cst[tid] + sigf(zi) * tanhf(zg);
      cst[tid] = cn;
      h0g[g * 8 + tid] = sigf(zo) * tanhf(cn);
    }
    grid_barrier(bar);   // h0 complete

    // ================= layer 1 =================
    for (int i = tid; i < K1; i += NTHR)
      vbuf[i] = (__bf16)((i < HID) ? h0g[i] : h1g[i - HID]);
    if (tid < 32) zred[tid] = 0.f;
    __syncthreads();
    {
      v8f acc = gemv_tile((const unsigned char*)wl1, vbuf, tile, NC1, part, lane);
      if ((lane & 15) == 0) {
        int ro = tile * 16 + ((lane >> 4) << 3);
#pragma unroll
        for (int i = 0; i < 8; ++i) atomicAdd(&zred[ro + i], acc[i]);
      }
    }
    __syncthreads();
    if (tid < 8) {
      float zi = zred[tid] + b1l[tid], zf = zred[8 + tid] + b1l[8 + tid];
      float zg = zred[16 + tid] + b1l[16 + tid], zo = zred[24 + tid] + b1l[24 + tid];
      float cn = sigf(zf) * cst[8 + tid] + sigf(zi) * tanhf(zg);
      cst[8 + tid] = cn;
      h1g[g * 8 + tid] = sigf(zo) * tanhf(cn);
    }
    grid_barrier(bar);   // h1 complete

    // ========== head: y, softmax, feedback embed (WG0 only) ==========
    if (g == 0) {
      for (int i = tid; i < HID; i += NTHR) vbuf[i] = (__bf16)h1g[i];
      __syncthreads();
      const unsigned char* wlg = ws + WS_WLIN;   // streamed from L2 each step
#pragma unroll
      for (int tt = 0; tt < 2; ++tt) {           // wave w -> tiles w and w+8
        int tau = w + tt * 8;
        // full-chunk tile: dual accumulators over even/odd chunks
        v8f a0 = {}, a1 = {};
        for (int c = 0; c < NCLIN; c += 2) {
          a0 = gemv_wmma(wlg, vbuf, tau * NCLIN, c, lane, a0);
          a1 = gemv_wmma(wlg, vbuf, tau * NCLIN, c + 1, lane, a1);
        }
        if ((lane & 15) == 0) {
          int r = tau * 16 + ((lane >> 4) << 3);
#pragma unroll
          for (int i = 0; i < 8; ++i) ybuf[r + i] = a0[i] + a1[i] + blin[r + i];
        }
      }
      __syncthreads();
      // softmax over 256 (wave-shuffle + LDS tree)
      float mv = ybuf[tid];
#pragma unroll
      for (int off = 16; off > 0; off >>= 1) mv = fmaxf(mv, __shfl_xor(mv, off, 32));
      if (lane == 0) wred[w] = mv;
      __syncthreads();
      if (tid == 0) {
        float m2 = wred[0];
        for (int i = 1; i < 8; ++i) m2 = fmaxf(m2, wred[i]);
        wred[8] = m2;
      }
      __syncthreads();
      float ev = __expf(ybuf[tid] - wred[8]);
      float sv = ev;
#pragma unroll
      for (int off = 16; off > 0; off >>= 1) sv += __shfl_xor(sv, off, 32);
      if (lane == 0) wred[16 + w] = sv;
      __syncthreads();
      if (tid == 0) {
        float s2 = 0.f;
        for (int i = 0; i < 8; ++i) s2 += wred[16 + i];
        wred[9] = 1.f / s2;
      }
      __syncthreads();
      float p = ev * wred[9];
      out[(long long)t * OUTD + tid] = p;        // output row t = softmax(y_t)
      pbuf[tid] = (__bf16)p;
      __syncthreads();
      {  // embed = W_map @ p + b_map : wave w -> tile w (no cross-wave reduce)
        const unsigned char* wmg = ws + WS_WMAP;
        v8f a0 = {}, a1 = {};
#pragma unroll
        for (int c = 0; c < NCMAP; c += 2) {
          a0 = gemv_wmma(wmg, pbuf, w * NCMAP, c, lane, a0);
          a1 = gemv_wmma(wmg, pbuf, w * NCMAP, c + 1, lane, a1);
        }
        if ((lane & 15) == 0) {
          int r = w * 16 + ((lane >> 4) << 3);
#pragma unroll
          for (int i = 0; i < 8; ++i) embg[r + i] = a0[i] + a1[i] + bmap[r + i];
        }
      }
    }
    grid_barrier(bar);   // feedback embed complete -> next step
  }
}

// ================= launch =================
extern "C" void kernel_launch(void* const* d_in, const int* in_sizes, int n_in,
                              void* d_out, int out_size, void* d_ws, size_t ws_size,
                              hipStream_t stream) {
  const float* x    = (const float*)d_in[0];
  const float* Wih0 = (const float*)d_in[1];
  const float* Whh0 = (const float*)d_in[2];
  const float* bih0 = (const float*)d_in[3];
  const float* bhh0 = (const float*)d_in[4];
  const float* Wih1 = (const float*)d_in[5];
  const float* Whh1 = (const float*)d_in[6];
  const float* bih1 = (const float*)d_in[7];
  const float* bhh1 = (const float*)d_in[8];
  const float* Wlin = (const float*)d_in[9];
  const float* blin = (const float*)d_in[10];
  const float* Wmap = (const float*)d_in[11];
  const float* bmap = (const float*)d_in[12];
  unsigned char* ws = (unsigned char*)d_ws;
  float* out = (float*)d_out;

  (void)hipFuncSetAttribute(reinterpret_cast<const void*>(lstm_persistent),
                            hipFuncAttributeMaxDynamicSharedMemorySize, (int)LDS_TOTAL);

  long long ntot = NP0 + NP1 + NPL + NPM;        // ~15.8M pack elements (+state zeroing)
  int pblocks = (int)((ntot + NTHR - 1) / NTHR);
  lstm_prep<<<pblocks, NTHR, 0, stream>>>(Wih0, Whh0, Wih1, Whh1, Wlin, Wmap, ws);
  lstm_persistent<<<NWG, NTHR, LDS_TOTAL, stream>>>(x, bih0, bhh0, bih1, bhh1, blin, bmap, ws, out);
}
